// SAttn_52956946760140
// MI455X (gfx1250) — matrix operand order repaired and checked by reading, not compiled, not run on hardware
//
#include <hip/hip_runtime.h>
#include <hip/hip_bf16.h>
#include <math.h>

typedef __attribute__((ext_vector_type(16))) _Float16 v16h;
typedef __attribute__((ext_vector_type(8)))  _Float16 v8h;
typedef __attribute__((ext_vector_type(8)))  float    v8f;
typedef __attribute__((ext_vector_type(4)))  float    v4f;

#define DEV static __device__ __forceinline__

constexpr int TOK = 4096;          // B*S
constexpr int DIMC = 512;
constexpr int NEL = TOK * DIMC;    // 2,097,152

// ---------------------------------------------------------------- WMMA utils
DEV v8f wmma32(v16h a, v16h b, v8f c) {
  return __builtin_amdgcn_wmma_f32_16x16x32_f16(false, a, false, b, (short)0, c,
                                                false, false);
}

DEV v16h cat8(v8h lo, v8h hi) {
  return __builtin_shufflevector(lo, hi, 0, 1, 2, 3, 4, 5, 6, 7, 8, 9, 10, 11,
                                 12, 13, 14, 15);
}

DEV void cp8(_Float16* dst, const _Float16* src) {
  *(v8h*)dst = *(const v8h*)src;
}

// A fragment 16(M)x32(K), row-major src[m*ld + k]; ld must be mult of 8 halfs.
DEV v16h frag_a(const _Float16* src, int ld, int lane) {
  int m = lane & 15, h = lane >> 4;
  const _Float16* r = src + m * ld + h * 16;
  return cat8(*(const v8h*)r, *(const v8h*)(r + 8));
}

// B fragment 32(K)x16(N) with B[k][n] = src[n*ld + k] (n-major source).
DEV v16h frag_b_tr(const _Float16* src, int ld, int lane) {
  int n = lane & 15, h = lane >> 4;
  const _Float16* r = src + n * ld + h * 16;
  return cat8(*(const v8h*)r, *(const v8h*)(r + 8));
}

// C/D 16x16 f32: lane l holds col n=l&15; reg r holds row m = r + 8*(l>>4)
DEV void store_c(float* dst, int ld, int lane, v8f c) {
  int n = lane & 15, h = lane >> 4;
#pragma unroll
  for (int r = 0; r < 8; ++r) dst[(r + 8 * h) * ld + n] = c[r];
}

// ---------------------------------------------------------------- kernels
__global__ void k_cvt_f16(const float* __restrict__ x, _Float16* __restrict__ y,
                          int n) {
  int i = blockIdx.x * blockDim.x + threadIdx.x;
  if (i < n) y[i] = (_Float16)x[i];
}

// f32 [rows,cols] -> f16 transposed [cols,rows]; blockIdx.y = batch (heads)
__global__ void k_cvt_tr(const float* __restrict__ x, _Float16* __restrict__ y,
                         int rows, int cols) {
  const float* xb = x + (size_t)blockIdx.y * rows * cols;
  _Float16* yb = y + (size_t)blockIdx.y * rows * cols;
  int i = blockIdx.x * blockDim.x + threadIdx.x;
  if (i < rows * cols) {
    int r = i / cols, c = i % cols;
    yb[(size_t)c * rows + r] = (_Float16)xb[i];
  }
}

// Y0 = X@W0, Y1 = X@W1 (Wt stored [n][k]). Block tile 128(M)x64(N).
// Waves 0-3 -> W0 rows (w&3)*32, waves 4-7 -> W1 (shared A tile).
// 8 WMMA per wave per k-step, operands batch-loaded before the WMMA burst.
__global__ void __launch_bounds__(256) k_dual_gemm(
    const _Float16* __restrict__ X, const _Float16* __restrict__ W0t,
    const _Float16* __restrict__ W1t, float* __restrict__ Y0,
    float* __restrict__ Y1) {
  constexpr int AP = 40, BP = 40;
  __shared__ alignas(16) _Float16 As[128 * AP];
  __shared__ alignas(16) _Float16 B0s[64 * BP];
  __shared__ alignas(16) _Float16 B1s[64 * BP];
  int n0 = blockIdx.x * 64, m0 = blockIdx.y * 128;
  int tid = threadIdx.x, lane = tid & 31, wave = tid >> 5;
  v8f acc[8] = {};
  for (int kk = 0; kk < DIMC; kk += 32) {
#pragma unroll
    for (int rep = 0; rep < 2; ++rep) {  // A 128x32
      int idx = (tid + rep * 256) * 8, r = idx >> 5, c = idx & 31;
      cp8(As + r * AP + c, X + (size_t)(m0 + r) * DIMC + kk + c);
    }
    {  // B tiles 64(n) x 32(k)
      int idx = tid * 8, r = idx >> 5, c = idx & 31;
      cp8(B0s + r * BP + c, W0t + (size_t)(n0 + r) * DIMC + kk + c);
      cp8(B1s + r * BP + c, W1t + (size_t)(n0 + r) * DIMC + kk + c);
    }
    __syncthreads();
    const _Float16* Bs = (wave < 4) ? B0s : B1s;
    const _Float16* Ab = As + (wave & 3) * 32 * AP;
    v16h a0 = frag_a(Ab, AP, lane);
    v16h a1 = frag_a(Ab + 16 * AP, AP, lane);
    v16h b[4];
#pragma unroll
    for (int nt = 0; nt < 4; ++nt) b[nt] = frag_b_tr(Bs + nt * 16 * BP, BP, lane);
#pragma unroll
    for (int nt = 0; nt < 4; ++nt) {
      acc[nt] = wmma32(a0, b[nt], acc[nt]);
      acc[4 + nt] = wmma32(a1, b[nt], acc[4 + nt]);
    }
    __syncthreads();
  }
  float* Y = (wave < 4) ? Y0 : Y1;
  int mr = m0 + (wave & 3) * 32;
#pragma unroll
  for (int nt = 0; nt < 4; ++nt) {
    store_c(Y + (size_t)mr * DIMC + n0 + nt * 16, DIMC, lane, acc[nt]);
    store_c(Y + (size_t)(mr + 16) * DIMC + n0 + nt * 16, DIMC, lane,
            acc[4 + nt]);
  }
}

// LayerNorm over 512; optional residual add (Res) and f32 sum output (Sum).
__global__ void __launch_bounds__(256) k_layernorm(
    const float* __restrict__ X, const float* __restrict__ Res,
    float* __restrict__ Sum, _Float16* __restrict__ Y,
    const float* __restrict__ gamma, const float* __restrict__ beta) {
  __shared__ float red[256];
  int row = blockIdx.x, tid = threadIdx.x;
  const float* x = X + (size_t)row * DIMC;
  float v0 = x[tid], v1 = x[tid + 256];
  if (Res) {
    v0 += Res[(size_t)row * DIMC + tid];
    v1 += Res[(size_t)row * DIMC + tid + 256];
  }
  if (Sum) {
    Sum[(size_t)row * DIMC + tid] = v0;
    Sum[(size_t)row * DIMC + tid + 256] = v1;
  }
  red[tid] = v0 + v1;
  __syncthreads();
  for (int o = 128; o > 0; o >>= 1) {
    if (tid < o) red[tid] += red[tid + o];
    __syncthreads();
  }
  float mu = red[0] * (1.0f / DIMC);
  __syncthreads();
  float d0 = v0 - mu, d1 = v1 - mu;
  red[tid] = d0 * d0 + d1 * d1;
  __syncthreads();
  for (int o = 128; o > 0; o >>= 1) {
    if (tid < o) red[tid] += red[tid + o];
    __syncthreads();
  }
  float inv = rsqrtf(red[0] * (1.0f / DIMC) + 1e-6f);
  Y[(size_t)row * DIMC + tid] = (_Float16)(d0 * inv * gamma[tid] + beta[tid]);
  Y[(size_t)row * DIMC + tid + 256] =
      (_Float16)(d1 * inv * gamma[tid + 256] + beta[tid + 256]);
}

// Per-head 64x64 projections with transposed weights Wt[h][e][d].
// blockIdx.x: [0,8)=Q, [8,16)=K, [16,24)=V; h=blockIdx.x&7.
// Block tile 256 tokens, 32 rows/wave. Q,K out [b,h,s,64]; V out [b,h,64,s].
__global__ void __launch_bounds__(256) k_qkv(
    const _Float16* __restrict__ NM, const _Float16* __restrict__ ND,
    const _Float16* __restrict__ WQt, const _Float16* __restrict__ WKt,
    const _Float16* __restrict__ WVt, _Float16* __restrict__ Q,
    _Float16* __restrict__ K, _Float16* __restrict__ Vt) {
  constexpr int XP = 72, WP = 72;
  __shared__ alignas(16) _Float16 Xs[256 * XP];
  __shared__ alignas(16) _Float16 Ws[64 * WP];   // [e][d]
  int sel = blockIdx.x >> 3, h = blockIdx.x & 7;
  int t0 = blockIdx.y * 256;
  const _Float16* X = (sel == 0) ? NM : ND;
  const _Float16* W =
      ((sel == 0) ? WQt : (sel == 1) ? WKt : WVt) + (size_t)h * 64 * 64;
  int tid = threadIdx.x, lane = tid & 31, wave = tid >> 5;
#pragma unroll
  for (int rep = 0; rep < 8; ++rep) {  // X tile 256x64 (head slice)
    int idx = (tid + rep * 256) * 8, r = idx >> 6, c = idx & 63;
    cp8(Xs + r * XP + c, X + (size_t)(t0 + r) * DIMC + h * 64 + c);
  }
#pragma unroll
  for (int rep = 0; rep < 2; ++rep) {  // Wt 64x64 (e-major)
    int idx = (tid + rep * 256) * 8, r = idx >> 6, c = idx & 63;
    cp8(Ws + r * WP + c, W + (size_t)r * 64 + c);
  }
  __syncthreads();
  v8f acc[8] = {};
  const _Float16* Ab = Xs + wave * 32 * XP;
#pragma unroll
  for (int ks = 0; ks < 2; ++ks) {
    v16h a0 = frag_a(Ab + ks * 32, XP, lane);
    v16h a1 = frag_a(Ab + 16 * XP + ks * 32, XP, lane);
    v16h b[4];
#pragma unroll
    for (int nt = 0; nt < 4; ++nt)
      b[nt] = frag_b_tr(Ws + nt * 16 * WP + ks * 32, WP, lane);
#pragma unroll
    for (int nt = 0; nt < 4; ++nt) {
      acc[nt] = wmma32(a0, b[nt], acc[nt]);
      acc[4 + nt] = wmma32(a1, b[nt], acc[4 + nt]);
    }
  }
  int n = lane & 15, hh = lane >> 4;
#pragma unroll
  for (int af = 0; af < 2; ++af)
#pragma unroll
    for (int nt = 0; nt < 4; ++nt)
#pragma unroll
      for (int r = 0; r < 8; ++r) {
        int t = t0 + wave * 32 + af * 16 + r + 8 * hh;
        int b = t >> 11, s = t & 2047;
        float v = acc[af * 4 + nt][r];
        if (sel == 2)  // V transposed: Vt[(b*8+h)*64 + e][s]
          Vt[((size_t)((b * 8 + h) * 64) + nt * 16 + n) * 2048 + s] =
              (_Float16)v;
        else {
          _Float16* O = (sel == 0) ? Q : K;
          O[((size_t)(b * 8 + h) * 2048 + s) * 64 + nt * 16 + n] = (_Float16)v;
        }
      }
}

// Flash attention. One block = one (b,h) x 128 queries (8 waves x 16 queries).
// Key tile = 64 per iteration. S^T = K.Q^T (keys on VGPR axis). O^T = V^T.P^T.
// 16 WMMA per iteration; all fragments batch-loaded before each WMMA burst.
__global__ void __launch_bounds__(256) k_attn(const _Float16* __restrict__ Q,
                                              const _Float16* __restrict__ K,
                                              const _Float16* __restrict__ Vt,
                                              float* __restrict__ O) {
  constexpr int KS = 72, VP = 72, PP = 72;
  __shared__ alignas(16) _Float16 Ks[64 * KS];       // [key][d]   64x64
  __shared__ alignas(16) _Float16 Vs[64 * VP];       // [d][key]   64x64
  __shared__ alignas(16) _Float16 Ps[8 * 16 * PP];   // per-wave [query][key]
  int bh = blockIdx.y;
  int tid = threadIdx.x, lane = tid & 31, wave = tid >> 5;
  const _Float16* qb = Q + (size_t)bh * 2048 * 64;
  const _Float16* kb = K + (size_t)bh * 2048 * 64;
  const _Float16* vb = Vt + (size_t)bh * 64 * 2048;
  int q0 = blockIdx.x * 128 + wave * 16;
  // Q^T B-fragments (B[k=d][n=query] = q[query][d]) live in regs all loop long
  v16h bq0 = frag_b_tr(qb + (size_t)q0 * 64, 64, lane);
  v16h bq1 = frag_b_tr(qb + (size_t)q0 * 64 + 32, 64, lane);
  v8f oacc[4] = {};
  float m_run = -1e30f, l_run = 0.0f;
  _Float16* pst = Ps + wave * 16 * PP;
  int ci = lane & 15, ch = lane >> 4;
  for (int kk = 0; kk < 2048; kk += 64) {
#pragma unroll
    for (int rep = 0; rep < 2; ++rep) {  // K tile 64x64 (key-major)
      int idx = (tid + rep * 256) * 8, r = idx >> 6, c = idx & 63;
      cp8(Ks + r * KS + c, kb + (size_t)(kk + r) * 64 + c);
    }
#pragma unroll
    for (int rep = 0; rep < 2; ++rep) {  // V tile 64x64 (d-major)
      int idx = (tid + rep * 256) * 8, r = idx >> 6, c = idx & 63;
      cp8(Vs + r * VP + c, vb + (size_t)r * 2048 + kk + c);
    }
    __syncthreads();
    // S^T: 64 keys (4 row-frags) x 16 queries, K=64 in two steps
    v8f st[4] = {};
    {
      v16h al[4];
#pragma unroll
      for (int j = 0; j < 4; ++j) al[j] = frag_a(Ks + j * 16 * KS, KS, lane);
#pragma unroll
      for (int j = 0; j < 4; ++j) st[j] = wmma32(al[j], bq0, st[j]);
#pragma unroll
      for (int j = 0; j < 4; ++j)
        al[j] = frag_a(Ks + j * 16 * KS + 32, KS, lane);
#pragma unroll
      for (int j = 0; j < 4; ++j) st[j] = wmma32(al[j], bq1, st[j]);
    }
    // online softmax per query column (lanes ci and ci+16 pair up)
    float tmax = st[0][0];
#pragma unroll
    for (int j = 0; j < 4; ++j)
#pragma unroll
      for (int r = 0; r < 8; ++r) tmax = fmaxf(tmax, st[j][r]);
    tmax = fmaxf(tmax, __shfl_xor(tmax, 16, 32));
    float m_new = fmaxf(m_run, tmax);
    float alpha = __expf(m_run - m_new);
    float psum = 0.0f;
#pragma unroll
    for (int j = 0; j < 4; ++j) {
      v8h pk;
#pragma unroll
      for (int r = 0; r < 8; ++r) {
        float p = __expf(st[j][r] - m_new);  // key = 16*j + 8*ch + r
        psum += p;
        pk[r] = (_Float16)p;
      }
      *(v8h*)(pst + ci * PP + 16 * j + 8 * ch) = pk;
    }
    psum += __shfl_xor(psum, 16, 32);
    l_run = l_run * alpha + psum;
    m_run = m_new;
#pragma unroll
    for (int f = 0; f < 4; ++f)
#pragma unroll
      for (int r = 0; r < 8; ++r) oacc[f][r] *= alpha;
    // O^T += V^T(16d x 64keys) @ P^T(64keys x 16q), two 32-key steps
    v16h bp0 = frag_b_tr(pst, PP, lane);
    v16h bp1 = frag_b_tr(pst + 32, PP, lane);
    {
      v16h av[4];
#pragma unroll
      for (int f = 0; f < 4; ++f) av[f] = frag_a(Vs + f * 16 * VP, VP, lane);
#pragma unroll
      for (int f = 0; f < 4; ++f) oacc[f] = wmma32(av[f], bp0, oacc[f]);
#pragma unroll
      for (int f = 0; f < 4; ++f)
        av[f] = frag_a(Vs + f * 16 * VP + 32, VP, lane);
#pragma unroll
      for (int f = 0; f < 4; ++f) oacc[f] = wmma32(av[f], bp1, oacc[f]);
    }
    __syncthreads();
  }
  float inv = 1.0f / l_run;
  int b = bh >> 3, h = bh & 7;
  float* ob = O + ((size_t)(b * 2048 + q0 + ci)) * DIMC + h * 64;
#pragma unroll
  for (int f = 0; f < 4; ++f) {
    v4f o0, o1;
#pragma unroll
    for (int r = 0; r < 4; ++r) {
      o0[r] = oacc[f][r] * inv;
      o1[r] = oacc[f][4 + r] * inv;
    }
    float* p = ob + f * 16 + 8 * ch;
    *(v4f*)p = o0;
    *(v4f*)(p + 4) = o1;
  }
}

// out = mod2 + elu(nf @ Wf + bf). Block tile 256x64, 32 rows/wave.
__global__ void __launch_bounds__(256) k_ffn(const _Float16* __restrict__ X,
                                             const _Float16* __restrict__ Wt,
                                             const float* __restrict__ bf,
                                             const float* __restrict__ mod2,
                                             float* __restrict__ out) {
  constexpr int AP = 40, BP = 40;
  __shared__ alignas(16) _Float16 Xs[256 * AP];
  __shared__ alignas(16) _Float16 Bs[64 * BP];  // [n][k]
  int n0 = blockIdx.x * 64, m0 = blockIdx.y * 256;
  int tid = threadIdx.x, lane = tid & 31, wave = tid >> 5;
  v8f acc[8] = {};
  for (int kk = 0; kk < DIMC; kk += 32) {
#pragma unroll
    for (int rep = 0; rep < 4; ++rep) {  // A 256x32
      int idx = (tid + rep * 256) * 8, r = idx >> 5, c = idx & 31;
      cp8(Xs + r * AP + c, X + (size_t)(m0 + r) * DIMC + kk + c);
    }
    {  // B 64(n) x 32(k)
      int idx = tid * 8, r = idx >> 5, c = idx & 31;
      cp8(Bs + r * BP + c, Wt + (size_t)(n0 + r) * DIMC + kk + c);
    }
    __syncthreads();
    const _Float16* Ab = Xs + wave * 32 * AP;
    v16h a0 = frag_a(Ab, AP, lane);
    v16h a1 = frag_a(Ab + 16 * AP, AP, lane);
    v16h b[4];
#pragma unroll
    for (int nt = 0; nt < 4; ++nt) b[nt] = frag_b_tr(Bs + nt * 16 * BP, BP, lane);
#pragma unroll
    for (int nt = 0; nt < 4; ++nt) {
      acc[nt] = wmma32(a0, b[nt], acc[nt]);
      acc[4 + nt] = wmma32(a1, b[nt], acc[4 + nt]);
    }
    __syncthreads();
  }
  int ci = lane & 15, ch = lane >> 4;
#pragma unroll
  for (int af = 0; af < 2; ++af)
#pragma unroll
    for (int nt = 0; nt < 4; ++nt)
#pragma unroll
      for (int r = 0; r < 8; ++r) {
        int row = m0 + wave * 32 + af * 16 + r + 8 * ch;
        int col = n0 + nt * 16 + ci;
        float x = acc[af * 4 + nt][r] + bf[col];
        float e = x > 0.0f ? x : expm1f(x);
        out[(size_t)row * DIMC + col] = mod2[(size_t)row * DIMC + col] + e;
      }
}

// ---------------------------------------------------------------- launch
extern "C" void kernel_launch(void* const* d_in, const int* in_sizes, int n_in,
                              void* d_out, int out_size, void* d_ws,
                              size_t ws_size, hipStream_t stream) {
  const float* mod = (const float*)d_in[0];
  const float* Wl = (const float*)d_in[1];
  const float* Wd = (const float*)d_in[2];
  const float* Wq = (const float*)d_in[3];
  const float* Wk = (const float*)d_in[4];
  const float* Wv = (const float*)d_in[5];
  const float* gamma = (const float*)d_in[6];
  const float* beta = (const float*)d_in[7];
  const float* Wf = (const float*)d_in[8];
  const float* bf = (const float*)d_in[9];
  float* out = (float*)d_out;

  char* ws = (char*)d_ws;
  size_t off = 0;
  auto take = [&](size_t bytes) -> char* {
    char* p = ws + off;
    off += (bytes + 255) & ~(size_t)255;
    return p;
  };
  _Float16* mod_h = (_Float16*)take((size_t)NEL * 2);
  _Float16* Wlt = (_Float16*)take((size_t)DIMC * DIMC * 2);
  _Float16* Wdt = (_Float16*)take((size_t)DIMC * DIMC * 2);
  _Float16* Wft = (_Float16*)take((size_t)DIMC * DIMC * 2);
  _Float16* Wqt = (_Float16*)take((size_t)8 * 64 * 64 * 2);
  _Float16* Wkt = (_Float16*)take((size_t)8 * 64 * 64 * 2);
  _Float16* Wvt = (_Float16*)take((size_t)8 * 64 * 64 * 2);
  float* mod_l = (float*)take((size_t)NEL * 4);
  float* dec_l = (float*)take((size_t)NEL * 4);
  _Float16* nm = (_Float16*)take((size_t)NEL * 2);
  _Float16* nd = (_Float16*)take((size_t)NEL * 2);
  _Float16* qh = (_Float16*)take((size_t)NEL * 2);
  _Float16* kh = (_Float16*)take((size_t)NEL * 2);
  _Float16* vt = (_Float16*)take((size_t)NEL * 2);
  float* o_f = (float*)take((size_t)NEL * 4);
  float* mod2 = (float*)take((size_t)NEL * 4);
  _Float16* nf = (_Float16*)take((size_t)NEL * 2);
  (void)ws_size; (void)in_sizes; (void)n_in; (void)out_size;

  k_cvt_f16<<<(NEL + 255) / 256, 256, 0, stream>>>(mod, mod_h, NEL);
  {
    dim3 g((DIMC * DIMC + 255) / 256, 1);
    k_cvt_tr<<<g, 256, 0, stream>>>(Wl, Wlt, DIMC, DIMC);
    k_cvt_tr<<<g, 256, 0, stream>>>(Wd, Wdt, DIMC, DIMC);
    k_cvt_tr<<<g, 256, 0, stream>>>(Wf, Wft, DIMC, DIMC);
  }
  {
    dim3 g((64 * 64 + 255) / 256, 8);
    k_cvt_tr<<<g, 256, 0, stream>>>(Wq, Wqt, 64, 64);
    k_cvt_tr<<<g, 256, 0, stream>>>(Wk, Wkt, 64, 64);
    k_cvt_tr<<<g, 256, 0, stream>>>(Wv, Wvt, 64, 64);
  }

  k_dual_gemm<<<dim3(8, 32), 256, 0, stream>>>(mod_h, Wlt, Wdt, mod_l, dec_l);
  k_layernorm<<<TOK, 256, 0, stream>>>(mod_l, nullptr, nullptr, nm, gamma, beta);
  k_layernorm<<<TOK, 256, 0, stream>>>(dec_l, nullptr, nullptr, nd, gamma, beta);
  k_qkv<<<dim3(24, 16), 256, 0, stream>>>(nm, nd, Wqt, Wkt, Wvt, qh, kh, vt);
  k_attn<<<dim3(16, 16), 256, 0, stream>>>(qh, kh, vt, o_f);
  k_layernorm<<<TOK, 256, 0, stream>>>(mod_l, o_f, mod2, nf, gamma, beta);
  k_ffn<<<dim3(8, 16), 256, 0, stream>>>(nf, Wft, bf, mod2, out);
}
